// ESPI_MSG_MODEL_65197603553511
// MI455X (gfx1250) — matrix operand order repaired and checked by reading, not compiled, not run on hardware
//
#include <hip/hip_runtime.h>
#include <hip/hip_bf16.h>

#define N_NODES    100000
#define N_EDGES    1600000
#define HIDDEN     128
#define NUM_GRAPHS 128
#define N_TILES    (N_NODES / 16)   // 6250, exact
#define LDSTRIDE   (HIDDEN + 8)     // padded LDS row stride (shorts) -> breaks bank conflicts

typedef __attribute__((ext_vector_type(16))) __bf16          v16bf;
typedef __attribute__((ext_vector_type(16))) unsigned short  v16u16;
typedef __attribute__((ext_vector_type(8)))  unsigned short  v8u16;
typedef __attribute__((ext_vector_type(8)))  float           v8f;

// ---------------------------------------------------------------------------
// helpers
// ---------------------------------------------------------------------------
__device__ __forceinline__ float sigmoidf_(float x) {
    return 1.0f / (1.0f + __expf(-x));
}

// Load a 16x32 bf16 A-fragment from a row-major f32 matrix (row stride 128).
// ISA 16-bit A 16x32 layout: lanes 0-15 -> M=lane, elems 0..7 = K0..7,
// elems 8..15 = K16..23; lanes 16-31 -> M=lane-16, K offset +8 / +24.
__device__ __forceinline__ v16bf load_a_frag(const float* __restrict__ base,
                                             int k0, int lane) {
    int m     = lane & 15;
    int khalf = (lane >> 4) * 8;
    const float* p = base + (size_t)m * HIDDEN + k0 + khalf;
    v16bf a;
#pragma unroll
    for (int i = 0; i < 8; ++i) a[i]     = (__bf16)p[i];
#pragma unroll
    for (int i = 0; i < 8; ++i) a[8 + i] = (__bf16)p[16 + i];
    return a;
}

// C += A(16x128) * Wtile(128x16) with bf16 weights staged in LDS
// (padded row stride LDSTRIDE shorts). B 32x16 layout: lanes 0-15 -> N=lane,
// K=0..7,16..23; lanes 16-31 -> N=lane-16, K=8..15,24..31.
__device__ __forceinline__ v8f mm_tile_lds(const v16bf a[4],
                                           const unsigned short* lw,
                                           int col0, int lane) {
    v8f c = {};
    int n    = lane & 15;
    int koff = (lane >> 4) * 8;
    const unsigned short* p = lw + (col0 + n) * LDSTRIDE + koff;
#pragma unroll
    for (int ks = 0; ks < 4; ++ks) {
        v8u16 lo = *(const v8u16*)(p + ks * 32);
        v8u16 hi = *(const v8u16*)(p + ks * 32 + 16);
        v16u16 raw = __builtin_shufflevector(lo, hi, 0, 1, 2, 3, 4, 5, 6, 7,
                                             8, 9, 10, 11, 12, 13, 14, 15);
        v16bf b = __builtin_bit_cast(v16bf, raw);
        c = __builtin_amdgcn_wmma_f32_16x16x32_bf16(false, a[ks], false, b,
                                                    (short)0, c, false, false);
    }
    return c;
}

// cooperative copy of a [rows x 128] bf16 matrix into padded LDS
__device__ __forceinline__ void stage_weights(const unsigned short* __restrict__ g,
                                              unsigned short* l, int rows,
                                              int tid, int nthreads) {
    for (int c = tid; c < rows * 16; c += nthreads) {
        int row = c >> 4, ch = c & 15;
        *(v8u16*)(l + row * LDSTRIDE + ch * 8) =
            *(const v8u16*)(g + (size_t)row * HIDDEN + ch * 8);
    }
}

__device__ __forceinline__ void atomicMaxFloat(float* addr, float val) {
    if (val >= 0.0f) atomicMax((int*)addr, __float_as_int(val));
    else             atomicMin((unsigned int*)addr, __float_as_uint(val));
}

// ---------------------------------------------------------------------------
// kernels
// ---------------------------------------------------------------------------
__global__ void cvt_bf16_kernel(const float* __restrict__ src,
                                unsigned short* __restrict__ dst, int count) {
    int i = blockIdx.x * blockDim.x + threadIdx.x;
    if (i < count) {
        __bf16 b = (__bf16)src[i];
        dst[i] = __builtin_bit_cast(unsigned short, b);
    }
}

__global__ void fill_kernel(float* __restrict__ p, float v, int count4) {
    int i = blockIdx.x * blockDim.x + threadIdx.x;
    if (i < count4) {
        float4 f = make_float4(v, v, v, v);
        *(float4*)(p + (size_t)i * 4) = f;
    }
}

__global__ void gather_kernel(const int* __restrict__ x,
                              const float* __restrict__ emb,
                              float* __restrict__ h) {
    int t = blockIdx.x * blockDim.x + threadIdx.x;        // (node, dword4)
    int node = t >> 5;
    if (node >= N_NODES) return;
    int c4 = (t & 31) * 4;
    float4 v = *(const float4*)(emb + (size_t)x[node] * HIDDEN + c4);
    *(float4*)(h + (size_t)node * HIDDEN + c4) = v;
}

// one wave per edge: 32 lanes x float4 = 128-float feature row
__global__ void scatter_add_kernel(const int* __restrict__ ei,
                                   const float* __restrict__ h,
                                   float* __restrict__ agg) {
    int wave = threadIdx.x >> 5;
    int lane = threadIdx.x & 31;
    int e = blockIdx.x * (blockDim.x >> 5) + wave;
    if (e >= N_EDGES) return;
    int s = ei[e];
    int d = ei[N_EDGES + e];
    float4 v = *(const float4*)(h + (size_t)s * HIDDEN + lane * 4);
    float* ap = agg + (size_t)d * HIDDEN + lane * 4;
    atomicAdd(ap + 0, v.x);
    atomicAdd(ap + 1, v.y);
    atomicAdd(ap + 2, v.z);
    atomicAdd(ap + 3, v.w);
}

// fused GRUCell: one wave per 16 nodes; bf16 Wih+Whh staged in 204KB of LDS
// (dynamic alloc; gfx1250 WGP has 320KB). Gates computed sequentially
// (r -> z -> n) to keep only 2 accumulators live and avoid spills.
__global__ void __launch_bounds__(256, 1)
gru_kernel(const float* __restrict__ agg,
           float* __restrict__ h,
           const unsigned short* __restrict__ Wih_bf,
           const unsigned short* __restrict__ Whh_bf,
           const float* __restrict__ bih,
           const float* __restrict__ bhh) {
    extern __shared__ __align__(32) unsigned short smem[];
    unsigned short* ldsWih = smem;
    unsigned short* ldsWhh = smem + 3 * HIDDEN * LDSTRIDE;

    stage_weights(Wih_bf, ldsWih, 3 * HIDDEN, threadIdx.x, blockDim.x);
    stage_weights(Whh_bf, ldsWhh, 3 * HIDDEN, threadIdx.x, blockDim.x);
    __syncthreads();

    int wave = threadIdx.x >> 5;
    int lane = threadIdx.x & 31;
    int tile = blockIdx.x * (blockDim.x >> 5) + wave;
    if (tile >= N_TILES) return;
    int row0 = tile * 16;

    v16bf Aagg[4], Ah[4];
#pragma unroll
    for (int ks = 0; ks < 4; ++ks) {
        Aagg[ks] = load_a_frag(agg + (size_t)row0 * HIDDEN, ks * 32, lane);
        Ah[ks]   = load_a_frag(h   + (size_t)row0 * HIDDEN, ks * 32, lane);
    }

    int n    = lane & 15;
    int mrow = 8 * (lane >> 4);

#pragma unroll 1
    for (int jr = 0; jr < 8; ++jr) {
        int colr = jr * 16;                // r-gate cols == hidden dim slice
        int colz = HIDDEN + colr;
        int coln = 2 * HIDDEN + colr;

        // r gate
        v8f t0 = mm_tile_lds(Aagg, ldsWih, colr, lane);
        v8f t1 = mm_tile_lds(Ah,   ldsWhh, colr, lane);
        float bir = bih[colr + n], bhr = bhh[colr + n];
        v8f r;
#pragma unroll
        for (int v = 0; v < 8; ++v) r[v] = sigmoidf_(t0[v] + bir + t1[v] + bhr);

        // z gate
        t0 = mm_tile_lds(Aagg, ldsWih, colz, lane);
        t1 = mm_tile_lds(Ah,   ldsWhh, colz, lane);
        float biz = bih[colz + n], bhz = bhh[colz + n];
        v8f z;
#pragma unroll
        for (int v = 0; v < 8; ++v) z[v] = sigmoidf_(t0[v] + biz + t1[v] + bhz);

        // n gate + state update (in-place: wave only touches its own rows)
        t0 = mm_tile_lds(Aagg, ldsWih, coln, lane);
        t1 = mm_tile_lds(Ah,   ldsWhh, coln, lane);
        float bin = bih[coln + n], bhn = bhh[coln + n];
#pragma unroll
        for (int v = 0; v < 8; ++v) {
            int row = row0 + v + mrow;
            float* hp = h + (size_t)row * HIDDEN + colr + n;
            float hold = *hp;
            float nn = tanhf(t0[v] + bin + r[v] * (t1[v] + bhn));
            *hp = (1.0f - z[v]) * nn + z[v] * hold;
        }
    }
}

// out = h @ dense_W.T + dense_b  (128x128 weight staged in 35KB static LDS)
__global__ void __launch_bounds__(256, 1)
dense_kernel(const float* __restrict__ h,
             const unsigned short* __restrict__ W_bf,
             const float* __restrict__ bias,
             float* __restrict__ out) {
    __shared__ __align__(32) unsigned short sdW[HIDDEN * LDSTRIDE];
    stage_weights(W_bf, sdW, HIDDEN, threadIdx.x, blockDim.x);
    __syncthreads();

    int wave = threadIdx.x >> 5;
    int lane = threadIdx.x & 31;
    int tile = blockIdx.x * (blockDim.x >> 5) + wave;
    if (tile >= N_TILES) return;
    int row0 = tile * 16;

    v16bf Ah[4];
#pragma unroll
    for (int ks = 0; ks < 4; ++ks)
        Ah[ks] = load_a_frag(h + (size_t)row0 * HIDDEN, ks * 32, lane);

    int n    = lane & 15;
    int mrow = 8 * (lane >> 4);

#pragma unroll 1
    for (int jc = 0; jc < 8; ++jc) {
        int col0 = jc * 16;
        v8f c = mm_tile_lds(Ah, sdW, col0, lane);
        float b = bias[col0 + n];
#pragma unroll
        for (int v = 0; v < 8; ++v) {
            int row = row0 + v + mrow;
            out[(size_t)row * HIDDEN + col0 + n] = c[v] + b;
        }
    }
}

__global__ void pool_max_kernel(const float* __restrict__ hd,
                                const int* __restrict__ batch,
                                float* __restrict__ g) {
    int t = blockIdx.x * blockDim.x + threadIdx.x;
    int node = t >> 5;
    if (node >= N_NODES) return;
    int c4 = (t & 31) * 4;
    int gi = batch[node];
    float4 v = *(const float4*)(hd + (size_t)node * HIDDEN + c4);
    float* gp = g + (size_t)gi * HIDDEN + c4;
    atomicMaxFloat(gp + 0, v.x);
    atomicMaxFloat(gp + 1, v.y);
    atomicMaxFloat(gp + 2, v.z);
    atomicMaxFloat(gp + 3, v.w);
}

__global__ void clf_kernel(const float* __restrict__ g,
                           const float* __restrict__ W,
                           const float* __restrict__ b,
                           float* __restrict__ out) {
    int i = threadIdx.x;          // 128 graphs
    float acc = b[0];
#pragma unroll 4
    for (int d = 0; d < HIDDEN; ++d) acc += g[(size_t)i * HIDDEN + d] * W[d];
    out[i] = sigmoidf_(acc);
}

// ---------------------------------------------------------------------------
// launch
// ---------------------------------------------------------------------------
extern "C" void kernel_launch(void* const* d_in, const int* in_sizes, int n_in,
                              void* d_out, int out_size, void* d_ws, size_t ws_size,
                              hipStream_t stream) {
    const int*   x       = (const int*)d_in[0];
    const int*   ei      = (const int*)d_in[1];
    const int*   batch   = (const int*)d_in[2];
    const float* emb     = (const float*)d_in[3];
    const float* Wih0    = (const float*)d_in[4];
    const float* Whh0    = (const float*)d_in[5];
    const float* bih0    = (const float*)d_in[6];
    const float* bhh0    = (const float*)d_in[7];
    const float* Wih1    = (const float*)d_in[8];
    const float* Whh1    = (const float*)d_in[9];
    const float* bih1    = (const float*)d_in[10];
    const float* bhh1    = (const float*)d_in[11];
    const float* dense_W = (const float*)d_in[12];
    const float* dense_b = (const float*)d_in[13];
    const float* clf_W   = (const float*)d_in[14];
    const float* clf_b   = (const float*)d_in[15];
    float* out = (float*)d_out;

    // workspace layout (all offsets 256B aligned)
    const size_t HB = (size_t)N_NODES * HIDDEN * sizeof(float);  // 51,200,000
    char* ws = (char*)d_ws;
    float*          h        = (float*)(ws);
    float*          agg      = (float*)(ws + HB);
    unsigned short* Wih0_bf  = (unsigned short*)(ws + 2 * HB);
    unsigned short* Whh0_bf  = Wih0_bf + 3 * HIDDEN * HIDDEN;
    unsigned short* Wih1_bf  = Whh0_bf + 3 * HIDDEN * HIDDEN;
    unsigned short* Whh1_bf  = Wih1_bf + 3 * HIDDEN * HIDDEN;
    unsigned short* dW_bf    = Whh1_bf + 3 * HIDDEN * HIDDEN;
    float*          g        = (float*)(dW_bf + HIDDEN * HIDDEN);

    const int B = 256;
    const int W3H = 3 * HIDDEN * HIDDEN;   // 49152
    const int WH  = HIDDEN * HIDDEN;       // 16384
    const size_t GRU_LDS = (size_t)2 * 3 * HIDDEN * LDSTRIDE * sizeof(unsigned short); // 208,896

    // allow >64KB dynamic LDS for the GRU kernel (gfx1250: 320KB per WGP)
    (void)hipFuncSetAttribute((const void*)gru_kernel,
                              hipFuncAttributeMaxDynamicSharedMemorySize,
                              (int)GRU_LDS);

    // weight f32 -> bf16 conversion
    cvt_bf16_kernel<<<(W3H + B - 1) / B, B, 0, stream>>>(Wih0, Wih0_bf, W3H);
    cvt_bf16_kernel<<<(W3H + B - 1) / B, B, 0, stream>>>(Whh0, Whh0_bf, W3H);
    cvt_bf16_kernel<<<(W3H + B - 1) / B, B, 0, stream>>>(Wih1, Wih1_bf, W3H);
    cvt_bf16_kernel<<<(W3H + B - 1) / B, B, 0, stream>>>(Whh1, Whh1_bf, W3H);
    cvt_bf16_kernel<<<(WH + B - 1) / B, B, 0, stream>>>(dense_W, dW_bf, WH);

    // h = emb[x]
    int gth = N_NODES * 32;
    gather_kernel<<<(gth + B - 1) / B, B, 0, stream>>>(x, emb, h);

    int scatter_blocks = (N_EDGES + (B / 32) - 1) / (B / 32);   // 200000
    int tile_blocks    = (N_TILES + (B / 32) - 1) / (B / 32);   // 782
    int fill_agg4      = N_NODES * HIDDEN / 4;

    // ---- layer 0 ----
    fill_kernel<<<(fill_agg4 + B - 1) / B, B, 0, stream>>>(agg, 0.0f, fill_agg4);
    scatter_add_kernel<<<scatter_blocks, B, 0, stream>>>(ei, h, agg);
    gru_kernel<<<tile_blocks, B, GRU_LDS, stream>>>(agg, h, Wih0_bf, Whh0_bf, bih0, bhh0);

    // ---- layer 1 ----
    fill_kernel<<<(fill_agg4 + B - 1) / B, B, 0, stream>>>(agg, 0.0f, fill_agg4);
    scatter_add_kernel<<<scatter_blocks, B, 0, stream>>>(ei, h, agg);
    gru_kernel<<<tile_blocks, B, GRU_LDS, stream>>>(agg, h, Wih1_bf, Whh1_bf, bih1, bhh1);

    // ---- dense (output into agg buffer) ----
    dense_kernel<<<tile_blocks, B, 0, stream>>>(h, dW_bf, dense_b, agg);

    // ---- global max pool ----
    int g4 = NUM_GRAPHS * HIDDEN / 4;
    fill_kernel<<<(g4 + B - 1) / B, B, 0, stream>>>(g, -INFINITY, g4);
    pool_max_kernel<<<(gth + B - 1) / B, B, 0, stream>>>(agg, batch, g);

    // ---- classifier ----
    clf_kernel<<<1, NUM_GRAPHS, 0, stream>>>(g, clf_W, clf_b, out);
    (void)in_sizes; (void)n_in; (void)out_size; (void)ws_size;
}
